// AttnBlock_17042430230963
// MI455X (gfx1250) — matrix operand order repaired and checked by reading, not compiled, not run on hardware
//
#include <hip/hip_runtime.h>

// ---------------------------------------------------------------------------
// Linear-attention block (B=8, N=4096, C=512, H=8, Dh=64) for gfx1250.
// Pipeline: LN1 -> f16 ; QKV GEMM (WMMA f16, double-buffered) ; q row-softmax;
// fused k col-softmax + context (K^T V) ; Q@context (WMMA, per-(b,h) B) ;
// out-proj GEMM (WMMA) ; LN2 + bias + residual.
// ---------------------------------------------------------------------------

typedef __attribute__((ext_vector_type(16))) _Float16 v16h;
typedef __attribute__((ext_vector_type(8)))  _Float16 v8h;
typedef __attribute__((ext_vector_type(8)))  float    v8f;
typedef __attribute__((ext_vector_type(4)))  float    v4f;

#define BB    8
#define NN    4096
#define CC    512
#define HH    8
#define DH    64
#define BN    (BB * NN)      // 32768 rows
#define F3    (3 * CC)       // 1536 qkv cols

__device__ __forceinline__ float wred_sum(float v) {
#pragma unroll
  for (int m = 16; m >= 1; m >>= 1) v += __shfl_xor(v, m, 32);
  return v;
}
__device__ __forceinline__ float wred_max(float v) {
#pragma unroll
  for (int m = 16; m >= 1; m >>= 1) v = fmaxf(v, __shfl_xor(v, m, 32));
  return v;
}

// ---------------- LayerNorm(x) -> f16 row-major (wave per 512-col row) -----
__global__ __launch_bounds__(256) void k_ln1(const float* __restrict__ x,
                                             const float* __restrict__ scale,
                                             _Float16* __restrict__ out) {
  const int wave = threadIdx.x >> 5, lane = threadIdx.x & 31;
  const long row = (long)blockIdx.x * 8 + wave;
  const float* xr = x + row * CC + lane * 16;
  float v[16];
  float s = 0.f, ss = 0.f;
#pragma unroll
  for (int i = 0; i < 4; i++) {
    v4f t = *(const v4f*)(xr + 4 * i);
#pragma unroll
    for (int j = 0; j < 4; j++) { float f = t[j]; v[4*i+j] = f; s += f; ss += f * f; }
  }
  s = wred_sum(s); ss = wred_sum(ss);
  const float mu = s * (1.f / CC);
  const float rstd = rsqrtf(ss * (1.f / CC) - mu * mu + 1e-5f);
  const float* sc = scale + lane * 16;
  _Float16* op = out + row * CC + lane * 16;
  v8h o0, o1;
#pragma unroll
  for (int i = 0; i < 8; i++) o0[i] = (_Float16)((v[i]     - mu) * rstd * sc[i]);
#pragma unroll
  for (int i = 0; i < 8; i++) o1[i] = (_Float16)((v[8 + i] - mu) * rstd * sc[8 + i]);
  *(v8h*)(op)     = o0;
  *(v8h*)(op + 8) = o1;
}

// ---------------- pack f32 weight (K x N) into WMMA B-fragment order -------
// frag = kt*NT + nt ; within frag: lane (N=nt*16+(lane&15)), elem e
// (K = kt*32 + (lane>>4)*16 + e) -> 16 contiguous f16 per lane.
__global__ __launch_bounds__(256) void k_pack_b(const float* __restrict__ W,
                                                _Float16* __restrict__ out,
                                                int N, int total) {
  int p = blockIdx.x * 256 + threadIdx.x;
  if (p >= total) return;
  const int e = p & 15, lane = (p >> 4) & 31, frag = p >> 9;
  const int NT = N >> 4;
  const int kt = frag / NT, nt = frag - kt * NT;
  const int K  = kt * 32 + (lane >> 4) * 16 + e;
  const int Nc = nt * 16 + (lane & 15);
  out[p] = (_Float16)W[(long)K * N + Nc];
}

// ---------------- pack per-(b,h) context (64x64 f32) into B fragments ------
__global__ __launch_bounds__(256) void k_pack_ctx(const float* __restrict__ ctx,
                                                  _Float16* __restrict__ out) {
  const int p = blockIdx.x * 256 + threadIdx.x;       // 64*4096 total
  const int e = p & 15, lane = (p >> 4) & 31;
  const int frag = (p >> 9) & 7, bh = p >> 12;
  const int kt = frag >> 2, nt = frag & 3;
  const int K  = kt * 32 + (lane >> 4) * 16 + e;      // d index
  const int Nc = nt * 16 + (lane & 15);               // e index
  out[p] = (_Float16)ctx[(long)bh * 4096 + K * 64 + Nc];
}

// ---------------- WMMA GEMM: A (M x Kdim, f16 row-major) x Bpack -----------
// Block = 8 waves; wave -> 16x64 tile (4 x v8f accum); grid (M/128, Nout/64).
// K-loop is register double-buffered (2 K-steps/iter) so fragment loads for
// step kt+1 are in flight while the 4 WMMAs of step kt execute.
// PER_BH: B fragments indexed by (b = m0/4096, h = blockIdx.y), KT=2, NT=4.
template <bool OUT16, bool PER_BH>
__global__ __launch_bounds__(256) void k_gemm(const _Float16* __restrict__ A,
                                              const _Float16* __restrict__ Bp,
                                              void* __restrict__ out,
                                              int Kdim, int Nout) {
  const int wave = threadIdx.x >> 5, lane = threadIdx.x & 31;
  const int r = lane & 15, half = lane >> 4;
  const long m0 = (long)blockIdx.x * 128 + wave * 16;
  const int n0 = blockIdx.y * 64;
  const int KT = Kdim >> 5, NT = Nout >> 4;

  const _Float16* arow = A + (m0 + r) * Kdim + half * 16;
  const int bsel = PER_BH ? ((int)(m0 >> 12) * HH + blockIdx.y) : 0;

  auto ldA = [&](int kt) -> v16h {
    const v8h alo = *(const v8h*)(arow + kt * 32);
    const v8h ahi = *(const v8h*)(arow + kt * 32 + 8);
    return __builtin_shufflevector(alo, ahi, 0, 1, 2, 3, 4, 5, 6, 7,
                                   8, 9, 10, 11, 12, 13, 14, 15);
  };
  auto bptr = [&](int kt) -> const _Float16* {
    if (PER_BH) return Bp + ((long)bsel * 8 + kt * 4) * 512 + lane * 16;
    else        return Bp + ((long)kt * NT + (n0 >> 4)) * 512 + lane * 16;
  };
  auto ldB = [&](const _Float16* bp, int j) -> v16h {
    const v8h blo = *(const v8h*)(bp + j * 512);
    const v8h bhi = *(const v8h*)(bp + j * 512 + 8);
    return __builtin_shufflevector(blo, bhi, 0, 1, 2, 3, 4, 5, 6, 7,
                                   8, 9, 10, 11, 12, 13, 14, 15);
  };

  const v8f zero = {0.f, 0.f, 0.f, 0.f, 0.f, 0.f, 0.f, 0.f};
  v8f acc[4];
#pragma unroll
  for (int j = 0; j < 4; j++) acc[j] = zero;

  // -------- prologue: fragments for kt = 0 --------
  v16h a0 = ldA(0);
  v16h b0[4];
  {
    const _Float16* bp = bptr(0);
#pragma unroll
    for (int j = 0; j < 4; j++) b0[j] = ldB(bp, j);
  }
  // -------- main loop: 2 K-steps per iteration (KT is even: 16 / 2 / 16) ---
  for (int kt = 0; kt < KT; kt += 2) {
    // issue loads for kt+1 while WMMAs of kt run
    v16h a1 = ldA(kt + 1);
    v16h b1[4];
    {
      const _Float16* bp = bptr(kt + 1);
#pragma unroll
      for (int j = 0; j < 4; j++) b1[j] = ldB(bp, j);
    }
#pragma unroll
    for (int j = 0; j < 4; j++)
      acc[j] = __builtin_amdgcn_wmma_f32_16x16x32_f16(
          false, a0, false, b0[j], (short)0, acc[j], false, false);
    if (kt + 2 < KT) {
      a0 = ldA(kt + 2);
      const _Float16* bp = bptr(kt + 2);
#pragma unroll
      for (int j = 0; j < 4; j++) b0[j] = ldB(bp, j);
    }
#pragma unroll
    for (int j = 0; j < 4; j++)
      acc[j] = __builtin_amdgcn_wmma_f32_16x16x32_f16(
          false, a1, false, b1[j], (short)0, acc[j], false, false);
  }

  // D layout: acc[j][v] -> row m0 + v + 8*half, col n0 + j*16 + r
  const long rowb = m0 + half * 8;
  if (OUT16) {
    _Float16* o = (_Float16*)out;
#pragma unroll
    for (int j = 0; j < 4; j++)
#pragma unroll
      for (int v = 0; v < 8; v++)
        o[(rowb + v) * Nout + n0 + j * 16 + r] = (_Float16)acc[j][v];
  } else {
    float* o = (float*)out;
#pragma unroll
    for (int j = 0; j < 4; j++)
#pragma unroll
      for (int v = 0; v < 8; v++)
        o[(rowb + v) * Nout + n0 + j * 16 + r] = acc[j][v];
  }
}

// ---------------- q softmax over d (wave per (row, head)), includes /8 -----
__global__ __launch_bounds__(256) void k_qsoftmax(const _Float16* __restrict__ qkv,
                                                  _Float16* __restrict__ q16) {
  const int wave = threadIdx.x >> 5, lane = threadIdx.x & 31;
  const long idx = (long)blockIdx.x * 8 + wave;   // over BN*H
  const long row = idx >> 3;
  const int h = (int)(idx & 7);
  const _Float16* p = qkv + row * F3 + h * DH + lane * 2;
  const float a = (float)p[0], b = (float)p[1];
  const float m = wred_max(fmaxf(a, b));
  const float ea = __expf(a - m), eb = __expf(b - m);
  const float s = wred_sum(ea + eb);
  const float inv = 1.f / (s * 8.f);              // softmax * 1/sqrt(64)
  _Float16* o = q16 + row * CC + h * DH + lane * 2;
  o[0] = (_Float16)(ea * inv);
  o[1] = (_Float16)(eb * inv);
}

// ---------------- fused k col-softmax + v/N + context = K^T V --------------
// One block per (b,h). ctx[bh][d][e] = sum_n softmax_n(k)[n,d] * v[n,e]/N.
__global__ __launch_bounds__(256) void k_context(const _Float16* __restrict__ qkv,
                                                 float* __restrict__ ctx) {
  __shared__ float part[4][64];
  __shared__ float smax[64], srsum[64];
  __shared__ float sk[32 * 64];
  __shared__ float sv[32 * 64];
  const int b = blockIdx.x >> 3, h = blockIdx.x & 7;
  const int t = threadIdx.x;
  const _Float16* kbase = qkv + (long)b * NN * F3 + CC + h * DH;
  const _Float16* vbase = kbase + CC;

  // pass 1: per-d max over n
  const int d = t & 63, qd = t >> 6;
  float m = -1e30f;
  for (int n = qd; n < NN; n += 4) m = fmaxf(m, (float)kbase[(long)n * F3 + d]);
  part[qd][d] = m;
  __syncthreads();
  if (t < 64)
    smax[t] = fmaxf(fmaxf(part[0][t], part[1][t]), fmaxf(part[2][t], part[3][t]));
  __syncthreads();
  // pass 2: per-d sum of exp
  const float mx = smax[d];
  float s = 0.f;
  for (int n = qd; n < NN; n += 4) s += __expf((float)kbase[(long)n * F3 + d] - mx);
  part[qd][d] = s;
  __syncthreads();
  if (t < 64) srsum[t] = 1.f / (part[0][t] + part[1][t] + part[2][t] + part[3][t]);
  __syncthreads();

  // pass 3: tiled accumulation; thread owns 4x4 of the 64x64 output
  float acc[4][4];
#pragma unroll
  for (int i = 0; i < 4; i++)
#pragma unroll
    for (int j = 0; j < 4; j++) acc[i][j] = 0.f;
  const int d0 = (t >> 4) * 4, e0 = (t & 15) * 4;
  const int ln = t >> 3, le = (t & 7) * 8;        // cooperative-load mapping
  for (int n0 = 0; n0 < NN; n0 += 32) {
    const _Float16* kr = kbase + (long)(n0 + ln) * F3 + le;
    const _Float16* vr = vbase + (long)(n0 + ln) * F3 + le;
#pragma unroll
    for (int i = 0; i < 8; i++) {
      sk[ln * 64 + le + i] = __expf((float)kr[i] - smax[le + i]) * srsum[le + i];
      sv[ln * 64 + le + i] = (float)vr[i] * (1.f / NN);
    }
    __syncthreads();
#pragma unroll 4
    for (int nn = 0; nn < 32; nn++) {
      float kd[4], ve[4];
#pragma unroll
      for (int i = 0; i < 4; i++) kd[i] = sk[nn * 64 + d0 + i];
#pragma unroll
      for (int j = 0; j < 4; j++) ve[j] = sv[nn * 64 + e0 + j];
#pragma unroll
      for (int i = 0; i < 4; i++)
#pragma unroll
        for (int j = 0; j < 4; j++) acc[i][j] = fmaf(kd[i], ve[j], acc[i][j]);
    }
    __syncthreads();
  }
  float* co = ctx + (long)blockIdx.x * 4096;
#pragma unroll
  for (int i = 0; i < 4; i++)
#pragma unroll
    for (int j = 0; j < 4; j++) co[(d0 + i) * 64 + e0 + j] = acc[i][j];
}

// ---------------- LN2(y + b_out) * scale + x (wave per row) ----------------
__global__ __launch_bounds__(256) void k_ln2(const float* __restrict__ Y,
                                             const float* __restrict__ bout,
                                             const float* __restrict__ scale,
                                             const float* __restrict__ x,
                                             float* __restrict__ out) {
  const int wave = threadIdx.x >> 5, lane = threadIdx.x & 31;
  const long row = (long)blockIdx.x * 8 + wave;
  const float* yr = Y + row * CC + lane * 16;
  const float* br = bout + lane * 16;
  float v[16];
  float s = 0.f, ss = 0.f;
#pragma unroll
  for (int i = 0; i < 16; i++) {
    const float f = yr[i] + br[i];
    v[i] = f; s += f; ss += f * f;
  }
  s = wred_sum(s); ss = wred_sum(ss);
  const float mu = s * (1.f / CC);
  const float rstd = rsqrtf(ss * (1.f / CC) - mu * mu + 1e-5f);
  const float* sc = scale + lane * 16;
  const float* xr = x + row * CC + lane * 16;
  float* o = out + row * CC + lane * 16;
#pragma unroll
  for (int i = 0; i < 16; i++) o[i] = (v[i] - mu) * rstd * sc[i] + xr[i];
}

// ---------------------------------------------------------------------------
extern "C" void kernel_launch(void* const* d_in, const int* in_sizes, int n_in,
                              void* d_out, int out_size, void* d_ws, size_t ws_size,
                              hipStream_t stream) {
  const float* x     = (const float*)d_in[0];
  const float* w_qkv = (const float*)d_in[1];
  const float* w_out = (const float*)d_in[2];
  const float* b_out = (const float*)d_in[3];
  const float* ln1s  = (const float*)d_in[4];
  const float* ln2s  = (const float*)d_in[5];
  float* out = (float*)d_out;

  char* ws = (char*)d_ws;
  size_t o = 0;
  _Float16* A16   = (_Float16*)(ws + o); o += (size_t)BN * CC * 2;   // LN1 out; reused as Q16
  _Float16* QKV16 = (_Float16*)(ws + o); o += (size_t)BN * F3 * 2;   // reused as Y32
  _Float16* ATT16 = (_Float16*)(ws + o); o += (size_t)BN * CC * 2;
  _Float16* WQP   = (_Float16*)(ws + o); o += (size_t)CC * F3 * 2;
  _Float16* WOP   = (_Float16*)(ws + o); o += (size_t)CC * CC * 2;
  float*    CTX   = (float*)   (ws + o); o += (size_t)BB * HH * DH * DH * 4;
  _Float16* CTXP  = (_Float16*)(ws + o); o += (size_t)BB * HH * DH * DH * 2;
  _Float16* Q16   = A16;            // A16 dead after QKV GEMM
  float*    Y32   = (float*)QKV16;  // QKV dead after qsoftmax + context

  // 1) pack weights into WMMA B-fragment order (f32 -> f16)
  k_pack_b<<<(CC * F3) / 256, 256, 0, stream>>>(w_qkv, WQP, F3, CC * F3);
  k_pack_b<<<(CC * CC) / 256, 256, 0, stream>>>(w_out, WOP, CC, CC * CC);
  // 2) LayerNorm1 -> f16
  k_ln1<<<BN / 8, 256, 0, stream>>>(x, ln1s, A16);
  // 3) QKV projection GEMM (32768x512 x 512x1536)
  k_gemm<true, false><<<dim3(BN / 128, F3 / 64), 256, 0, stream>>>(
      A16, WQP, QKV16, CC, F3);
  // 4) q softmax over d (and /sqrt(Dh))
  k_qsoftmax<<<(BN * HH) / 8, 256, 0, stream>>>(QKV16, Q16);
  // 5) fused k column-softmax + v/N + context = K^T V  (64 blocks)
  k_context<<<BB * HH, 256, 0, stream>>>(QKV16, CTX);
  // 6) pack context into per-(b,h) B fragments
  k_pack_ctx<<<(BB * HH * DH * DH) / 256, 256, 0, stream>>>(CTX, CTXP);
  // 7) out = Q @ context  (batched per (b,h) via PER_BH B selection)
  k_gemm<true, true><<<dim3(BN / 128, HH), 256, 0, stream>>>(
      Q16, CTXP, ATT16, DH, CC);
  // 8) output projection GEMM (32768x512 x 512x512) -> f32
  k_gemm<false, false><<<dim3(BN / 128, CC / 64), 256, 0, stream>>>(
      ATT16, WOP, Y32, CC, CC);
  // 9) LN2(+bias) * scale + residual
  k_ln2<<<BN / 8, 256, 0, stream>>>(Y32, b_out, ln2s, x, out);
}
